// InvariantPointAttention_65257733096117
// MI455X (gfx1250) — matrix operand order repaired and checked by reading, not compiled, not run on hardware
//
#include <hip/hip_runtime.h>
#include <hip/hip_bf16.h>

// ---------------- constants ----------------
#define NN 1024
#define HH 12
#define CC 16
#define CZ 128
#define PQ 4
#define PV 8
#define CS 384
#define COUT 2112   // H*(CZ + C + PV*4)
#define INFV 100000.0f

typedef __attribute__((ext_vector_type(16))) _Float16 v16h;
typedef __attribute__((ext_vector_type(8)))  _Float16 v8h;
typedef __attribute__((ext_vector_type(8)))  float    v8f;
typedef __attribute__((ext_vector_type(4)))  unsigned int v4u;
typedef __attribute__((ext_vector_type(8)))  int      v8i;
typedef __attribute__((ext_vector_type(4)))  int      v4i;

#ifndef __has_builtin
#define __has_builtin(x) 0
#endif
#if __has_builtin(__builtin_amdgcn_tensor_load_to_lds) && \
    __has_builtin(__builtin_amdgcn_s_wait_tensorcnt)
#define USE_TDM 1
#else
#define USE_TDM 0
#endif

#define CFENCE() asm volatile("" ::: "memory")

__device__ inline v8f wmma16(v16h a, v16h b, v8f c) {
  return __builtin_amdgcn_wmma_f32_16x16x32_f16(
      /*neg_a=*/false, a, /*neg_b=*/false, b,
      /*c_mod=*/(short)0, c, /*reuse_a=*/false, /*reuse_b=*/false);
}

// ---- fragment packers (f32 source -> f16 halves) ----
// A operand k order used by this code (16x32): half e of lane L:
//   K = (L>=16?16:0)+e, M = L&15
// B operand k order used by this code (32x16): half e of lane L:
//   K = (L>=16?16:0)+e, N = L&15

__device__ inline v16h packA_fast(const float* p) {  // p = row + k0 + kb (16B aligned)
  const float4* q = (const float4*)p;
  float4 a0 = q[0], a1 = q[1], b0 = q[2], b1 = q[3];
  v16h r;
  r[0]=(_Float16)a0.x; r[1]=(_Float16)a0.y; r[2]=(_Float16)a0.z; r[3]=(_Float16)a0.w;
  r[4]=(_Float16)a1.x; r[5]=(_Float16)a1.y; r[6]=(_Float16)a1.z; r[7]=(_Float16)a1.w;
  r[8]=(_Float16)b0.x; r[9]=(_Float16)b0.y; r[10]=(_Float16)b0.z; r[11]=(_Float16)b0.w;
  r[12]=(_Float16)b1.x; r[13]=(_Float16)b1.y; r[14]=(_Float16)b1.z; r[15]=(_Float16)b1.w;
  return r;
}

__device__ inline v16h packB_fast(const float* p) {  // 16 contiguous floats
  const float4* q = (const float4*)p;
  float4 a = q[0], b = q[1], c = q[2], d = q[3];
  v16h r;
  r[0]=(_Float16)a.x; r[1]=(_Float16)a.y; r[2]=(_Float16)a.z; r[3]=(_Float16)a.w;
  r[4]=(_Float16)b.x; r[5]=(_Float16)b.y; r[6]=(_Float16)b.z; r[7]=(_Float16)b.w;
  r[8]=(_Float16)c.x; r[9]=(_Float16)c.y; r[10]=(_Float16)c.z; r[11]=(_Float16)c.w;
  r[12]=(_Float16)d.x; r[13]=(_Float16)d.y; r[14]=(_Float16)d.z; r[15]=(_Float16)d.w;
  return r;
}

__device__ inline v16h packA_g(const float* row, int k0, int kb, int kmax, float scale) {
  v16h r;
#pragma unroll
  for (int e = 0; e < 16; ++e) {
    int kk = kb + e;
    int k = k0 + kk;
    r[e] = (_Float16)((k < kmax) ? row[k] * scale : 0.0f);
  }
  return r;
}

__device__ inline v16h packB_g(const float* row, int kstart, int kmax, float scale) {
  v16h r;
#pragma unroll
  for (int e = 0; e < 16; ++e) {
    int k = kstart + e;
    r[e] = (_Float16)((k < kmax) ? row[k] * scale : 0.0f);
  }
  return r;
}

#if USE_TDM
// Issue a TDM 2D tile load: 32 rows x 128 f32, row stride CZ, into LDS with
// pad_interval=128 dwords / pad_amount=4 dwords (i.e. LDS rows of 132 floats).
__device__ inline void tdm_load_tile(unsigned int lds_addr, const float* gptr) {
  unsigned long long ga = (unsigned long long)(uintptr_t)gptr;
  v4u g0;
  g0[0] = 1u;                                   // count=1, user desc
  g0[1] = lds_addr;                             // LDS byte address
  g0[2] = (unsigned int)(ga & 0xFFFFFFFFu);     // global_addr[31:0]
  g0[3] = (unsigned int)((ga >> 32) & 0x01FFFFFFu) | (2u << 30);  // addr[56:32] | type=2
  v8i g1;
  g1[0] = (2 << 16) | (1 << 20) | (6 << 22) | (3 << 25); // 4B data, pad_en, int=128dw, amt=4dw
  g1[1] = (int)(128u << 16);                    // tensor_dim0 = 128 (low 16 -> bits 79:64? packed)
  g1[2] = (int)(32u << 16);                     // tensor_dim0 hi=0, tensor_dim1 = 32
  g1[3] = (int)(128u << 16);                    // tensor_dim1 hi=0, tile_dim0 = 128
  g1[4] = 32;                                   // tile_dim1 = 32, tile_dim2 = 0
  g1[5] = 128;                                  // tensor_dim0_stride = CZ
  g1[6] = 0;
  g1[7] = 0;
  v4i gz = {0, 0, 0, 0};
#if __clang_major__ >= 23
  v8i gz8 = {0, 0, 0, 0, 0, 0, 0, 0};
  __builtin_amdgcn_tensor_load_to_lds(g0, g1, gz, gz, gz8, 0);
#else
  __builtin_amdgcn_tensor_load_to_lds(g0, g1, gz, gz, 0);
#endif
}
#endif

// ---------------- generic GEMM: out[M,Nc] = A[M,K] @ W[Nc,K]^T + bias ----------------
__global__ __launch_bounds__(32) void gemm_wmma(
    const float* __restrict__ A, int lda,
    const float* __restrict__ W, int ldw,
    const float* __restrict__ bias,
    float* __restrict__ out, int ldo, int K) {
  int lane = threadIdx.x & 31;
  int lh = lane & 15, hl = lane >> 4;
  int m0 = blockIdx.x * 16;
  int n0 = blockIdx.y * 16;
  const float* arow = A + (size_t)(m0 + lh) * lda + hl * 16;
  const float* wrow = W + (size_t)(n0 + lh) * ldw + hl * 16;
  v8f acc = {};
  for (int k0 = 0; k0 < K; k0 += 32) {
    __builtin_prefetch(arow + k0 + 128, 0, 0);
    v16h af = packA_fast(arow + k0);
    v16h bf = packB_fast(wrow + k0);
    acc = wmma16(af, bf, acc);
  }
  float b = bias ? bias[n0 + lh] : 0.0f;
#pragma unroll
  for (int r = 0; r < 8; ++r) {
    int M = r + 8 * hl;
    out[(size_t)(m0 + M) * ldo + n0 + lh] = acc[r] + b;
  }
}

// ---------------- apply frames to points, precompute point norms ----------------
__global__ void frames_kernel(const float* __restrict__ qp_proj,
                              const float* __restrict__ kvp_proj,
                              const float* __restrict__ rot,
                              const float* __restrict__ trans,
                              float* __restrict__ qpts, float* __restrict__ kpts,
                              float* __restrict__ vpts,
                              float* __restrict__ qn_ws, float* __restrict__ kn_ws) {
  int idx = blockIdx.x * blockDim.x + threadIdx.x;
  if (idx >= NN * HH) return;
  int i = idx / HH, h = idx % HH;
  float R[9], T[3];
#pragma unroll
  for (int t = 0; t < 9; ++t) R[t] = rot[(size_t)i * 9 + t];
#pragma unroll
  for (int t = 0; t < 3; ++t) T[t] = trans[(size_t)i * 3 + t];

  float qn = 0.0f;
#pragma unroll
  for (int p = 0; p < PQ; ++p) {
    float pl[3];
#pragma unroll
    for (int x = 0; x < 3; ++x) pl[x] = qp_proj[(size_t)i * (HH*PQ*3) + x * (HH*PQ) + h * PQ + p];
#pragma unroll
    for (int d = 0; d < 3; ++d) {
      float g = R[d*3+0]*pl[0] + R[d*3+1]*pl[1] + R[d*3+2]*pl[2] + T[d];
      qpts[(size_t)i * (HH*12) + h * 12 + p * 3 + d] = g;
      qn += g * g;
    }
  }
  qn_ws[h * NN + i] = qn;

  float kn = 0.0f;
#pragma unroll
  for (int p = 0; p < PQ + PV; ++p) {
    float pl[3];
#pragma unroll
    for (int x = 0; x < 3; ++x) pl[x] = kvp_proj[(size_t)i * (HH*12*3) + x * (HH*12) + h * 12 + p];
    float g[3];
#pragma unroll
    for (int d = 0; d < 3; ++d)
      g[d] = R[d*3+0]*pl[0] + R[d*3+1]*pl[1] + R[d*3+2]*pl[2] + T[d];
    if (p < PQ) {
#pragma unroll
      for (int d = 0; d < 3; ++d) kpts[(size_t)i * (HH*12) + h * 12 + p * 3 + d] = g[d];
      kn += g[0]*g[0] + g[1]*g[1] + g[2]*g[2];
    } else {
      int pv = p - PQ;
#pragma unroll
      for (int d = 0; d < 3; ++d) vpts[(size_t)i * (HH*24) + h * 24 + pv * 3 + d] = g[d];
    }
  }
  kn_ws[h * NN + i] = kn;
}

// ---------------- pair bias: bias_ws[h][i][j] = sqrt(1/3)*(z[i,j,:]@Wb[h,:] + bb[h]) ----------------
__global__ __launch_bounds__(32) void bias_kernel(
    const float* __restrict__ z, const float* __restrict__ Wb,
    const float* __restrict__ bb, float* __restrict__ bias_ws) {
  int lane = threadIdx.x & 31;
  int lh = lane & 15, hl = lane >> 4;
  int mt = blockIdx.x;                 // tile over flattened (i,j): N*N/16 tiles
  const float* arow = z + (size_t)(mt * 16 + lh) * CZ + hl * 16;
  const float* wrow = Wb + (size_t)lh * CZ;   // head row (guarded)
  int kmaxB = (lh < HH) ? CZ : 0;
  v16h bf[4];
#pragma unroll
  for (int kt = 0; kt < 4; ++kt) bf[kt] = packB_g(wrow, kt * 32 + hl * 16, kmaxB, 1.0f);
  v8f acc = {};
#pragma unroll
  for (int kt = 0; kt < 4; ++kt) {
    v16h af = packA_fast(arow + kt * 32);
    acc = wmma16(af, bf[kt], acc);
  }
  __builtin_prefetch(z + (size_t)(mt * 16 + 16 + lh) * CZ, 0, 0);
  const float s2 = 0.57735026918962576f;  // sqrt(1/3)
  if (lh < HH) {
    float b = bb[lh];
#pragma unroll
    for (int r = 0; r < 8; ++r) {
      int grow = mt * 16 + r + 8 * hl;
      int i = grow >> 10, j = grow & (NN - 1);
      bias_ws[((size_t)lh * NN + i) * NN + j] = s2 * (acc[r] + b);
    }
  }
}

// ---------------- fused attention per (head, 16-row query tile) ----------------
__global__ __launch_bounds__(32) void attn_kernel(
    const float* __restrict__ q_ws, const float* __restrict__ kv_ws,
    const float* __restrict__ qpts, const float* __restrict__ kpts,
    const float* __restrict__ vpts, const float* __restrict__ qn_ws,
    const float* __restrict__ kn_ws, const float* __restrict__ bias_ws,
    const float* __restrict__ mask, const float* __restrict__ head_w,
    _Float16* __restrict__ a_ws, float* __restrict__ cat,
    float* __restrict__ optraw) {
  __shared__ __align__(16) _Float16 Plds[16][1032];
  __shared__ float inv_l[16];
  int lane = threadIdx.x & 31;
  int lh = lane & 15, hl = lane >> 4;
  int h = blockIdx.x;
  int i0 = blockIdx.y * 16;

  float hw = log1pf(__expf(head_w[h])) * 0.13608276348795434f;  // softplus * sqrt(1/54)
  const float s1 = 0.14433756729740643f;                        // sqrt(1/(3*C))

  v16h qf  = packA_g(q_ws  + (size_t)(i0 + lh) * (HH*CC) + h * CC, 0, hl * 16, CC, s1);
  v16h qpf = packA_g(qpts  + (size_t)(i0 + lh) * (HH*12) + h * 12, 0, hl * 16, 12, hw);

  float qn_r[8], mi_r[8];
#pragma unroll
  for (int r = 0; r < 8; ++r) {
    int M = r + 8 * hl;
    qn_r[r] = qn_ws[h * NN + i0 + M];
    mi_r[r] = mask[i0 + M];
  }

  // ---- phase 1: logits into LDS (chained WMMAs) ----
  for (int jt = 0; jt < NN / 16; ++jt) {
    int j0 = jt * 16;
    float kn_j = kn_ws[h * NN + j0 + lh];
    float mj = mask[j0 + lh];
    v8f c0;
#pragma unroll
    for (int r = 0; r < 8; ++r) {
      int M = r + 8 * hl;
      float b = bias_ws[((size_t)h * NN + i0 + M) * NN + j0 + lh];
      c0[r] = b - 0.5f * hw * (qn_r[r] + kn_j) + INFV * (mi_r[r] * mj - 1.0f);
    }
    v16h kpf = packB_g(kpts + (size_t)(j0 + lh) * (HH*12) + h * 12, hl * 16, 12, 1.0f);
    v8f c1 = wmma16(qpf, kpf, c0);
    v16h kf = packB_g(kv_ws + (size_t)(j0 + lh) * (2*HH*CC) + h * CC, hl * 16, CC, 1.0f);
    v8f lg = wmma16(qf, kf, c1);
#pragma unroll
    for (int r = 0; r < 8; ++r) Plds[r + 8 * hl][j0 + lh] = (_Float16)lg[r];
  }
  __syncthreads();

  // ---- phase 2: softmax (vectorized 16B LDS accesses) ----
  {
    int row = lh, cbase = hl * 512;
    v8h* rp = (v8h*)&Plds[row][cbase];
    float mx = -1e30f;
    for (int t = 0; t < 64; ++t) {
      v8h x = rp[t];
#pragma unroll
      for (int e = 0; e < 8; ++e) mx = fmaxf(mx, (float)x[e]);
    }
    mx = fmaxf(mx, __shfl_xor(mx, 16, 32));
    float ssum = 0.0f;
    for (int t = 0; t < 64; ++t) {
      v8h x = rp[t];
      v8h y;
#pragma unroll
      for (int e = 0; e < 8; ++e) {
        float ev = __expf((float)x[e] - mx);
        ssum += ev;
        y[e] = (_Float16)ev;
      }
      rp[t] = y;
    }
    ssum += __shfl_xor(ssum, 16, 32);
    if (hl == 0) inv_l[row] = 1.0f / ssum;
  }
  __syncthreads();

  // ---- phase 2.5: store normalized attention (f16, 16B stores) ----
  for (int t = lane; t < 16 * (NN / 8); t += 32) {
    int row = t >> 7, j8 = (t & 127) * 8;
    v8h x = *(const v8h*)&Plds[row][j8];
    float inv = inv_l[row];
    v8h y;
#pragma unroll
    for (int e = 0; e < 8; ++e) y[e] = (_Float16)((float)x[e] * inv);
    *(v8h*)&a_ws[((size_t)h * NN + i0 + row) * NN + j8] = y;
  }

  // ---- phase 3: o and o_pt accumulation via WMMA ----
  v8f oacc = {}, op0 = {}, op1 = {};
  const int ldkv = 2 * HH * CC;  // 384
  for (int kt = 0; kt < NN / 32; ++kt) {
    int j0 = kt * 32;
    v16h pf;
#pragma unroll
    for (int e = 0; e < 8; ++e) pf[e] = Plds[lh][j0 + hl * 16 + e];
#pragma unroll
    for (int e = 8; e < 16; ++e) pf[e] = Plds[lh][j0 + hl * 16 + e];

    v16h vf, vp0, vp1;
    int vcol = HH * CC + h * CC + lh;  // v starts at col 192
#pragma unroll
    for (int e = 0; e < 16; ++e)
      vf[e] = (_Float16)kv_ws[(size_t)(j0 + hl * 16 + e) * ldkv + vcol];
    oacc = wmma16(pf, vf, oacc);

    int pc = h * 24;
#pragma unroll
    for (int e = 0; e < 16; ++e)
      vp0[e] = (_Float16)vpts[(size_t)(j0 + hl * 16 + e) * (HH*24) + pc + lh];
    op0 = wmma16(pf, vp0, op0);
    int n1 = 16 + lh;
#pragma unroll
    for (int e = 0; e < 16; ++e)
      vp1[e] = (n1 < 24) ? (_Float16)vpts[(size_t)(j0 + hl * 16 + e) * (HH*24) + pc + n1]
                         : (_Float16)0.0f;
    op1 = wmma16(pf, vp1, op1);
  }

#pragma unroll
  for (int r = 0; r < 8; ++r) {
    int M = r + 8 * hl;
    float inv = inv_l[M];
    cat[(size_t)(i0 + M) * COUT + h * CC + lh] = oacc[r] * inv;
    optraw[(size_t)(i0 + M) * (HH*24) + h * 24 + lh] = op0[r] * inv;
    if (16 + lh < 24)
      optraw[(size_t)(i0 + M) * (HH*24) + h * 24 + 16 + lh] = op1[r] * inv;
  }
}

// ---------------- o_pair per residue i: (heads x 128) = a[h,i,:] @ z[i,:,:] ----------------
// z tile staged via TDM (double-buffered) when available.
__global__ __launch_bounds__(32) void opair_kernel(
    const _Float16* __restrict__ a_ws, const float* __restrict__ z,
    float* __restrict__ cat) {
  __shared__ __align__(16) float zt[2][32][132];
  int lane = threadIdx.x & 31;
  int lh = lane & 15, hl = lane >> 4;
  int i = blockIdx.x;
  v8f acc[8];
#pragma unroll
  for (int nt = 0; nt < 8; ++nt) acc[nt] = v8f{};

  const _Float16* arow = a_ws + (size_t)lh * (NN * NN) + (size_t)i * NN;
  const float* zrow = z + (size_t)i * NN * CZ;

#if USE_TDM
  unsigned int lds_b[2];
  lds_b[0] = (unsigned int)(uintptr_t)(void*)&zt[0][0][0];
  lds_b[1] = (unsigned int)(uintptr_t)(void*)&zt[1][0][0];
  tdm_load_tile(lds_b[0], zrow);  // kt = 0
  for (int kt = 0; kt < NN / 32; ++kt) {
    int j0 = kt * 32;
    if (kt + 1 < NN / 32) {
      asm volatile("s_wait_dscnt 0x0" ::: "memory");   // prior reads of target buffer done
      tdm_load_tile(lds_b[(kt + 1) & 1], zrow + (size_t)(j0 + 32) * CZ);
      __builtin_amdgcn_s_wait_tensorcnt(1);            // current tile landed
    } else {
      __builtin_amdgcn_s_wait_tensorcnt(0);
    }
    CFENCE();
    float (*zb)[132] = zt[kt & 1];
#else
  for (int kt = 0; kt < NN / 32; ++kt) {
    int j0 = kt * 32;
    for (int t = lane; t < 32 * 32; t += 32) {
      int j = t >> 5, c4 = (t & 31) * 4;
      float4 v = *(const float4*)(zrow + (size_t)(j0 + j) * CZ + c4);
      zt[0][j][c4 + 0] = v.x; zt[0][j][c4 + 1] = v.y;
      zt[0][j][c4 + 2] = v.z; zt[0][j][c4 + 3] = v.w;
    }
    if (kt < NN / 32 - 1)
      __builtin_prefetch(zrow + (size_t)(j0 + 32 + (lane >> 2)) * CZ + (lane & 3) * 32, 0, 0);
    __syncthreads();
    float (*zb)[132] = zt[0];
#endif

    v16h af;
#pragma unroll
    for (int e = 0; e < 8; ++e) af[e] = arow[j0 + hl * 16 + e];
#pragma unroll
    for (int e = 8; e < 16; ++e) af[e] = arow[j0 + hl * 16 + e];

#pragma unroll
    for (int nt = 0; nt < 8; ++nt) {
      int c = nt * 16 + lh;
      v16h bf;
#pragma unroll
      for (int e = 0; e < 16; ++e) bf[e] = (_Float16)zb[hl * 16 + e][c];
      acc[nt] = wmma16(af, bf, acc[nt]);
    }
#if !USE_TDM
    __syncthreads();
#endif
  }

#pragma unroll
  for (int nt = 0; nt < 8; ++nt) {
#pragma unroll
    for (int r = 0; r < 8; ++r) {
      int M = r + 8 * hl;
      if (M < HH)
        cat[(size_t)i * COUT + HH*CC + 4*HH*PV + M * CZ + nt * 16 + lh] = acc[nt][r];
    }
  }
}

// ---------------- finalize points: inverse frame + norm into concat ----------------
__global__ void finalize_pts_kernel(const float* __restrict__ optraw,
                                    const float* __restrict__ rot,
                                    const float* __restrict__ trans,
                                    float* __restrict__ cat) {
  int idx = blockIdx.x * blockDim.x + threadIdx.x;
  if (idx >= NN * HH * PV) return;
  int i = idx / (HH * PV), pp = idx % (HH * PV);
  int h = pp / PV, pv = pp % PV;
  float g[3];
#pragma unroll
  for (int d = 0; d < 3; ++d)
    g[d] = optraw[(size_t)i * (HH*24) + h * 24 + pv * 3 + d] - trans[(size_t)i * 3 + d];
  float l[3];
#pragma unroll
  for (int d = 0; d < 3; ++d)
    l[d] = rot[(size_t)i*9 + 0*3 + d] * g[0] + rot[(size_t)i*9 + 1*3 + d] * g[1] +
           rot[(size_t)i*9 + 2*3 + d] * g[2];
  float nrm = sqrtf(l[0]*l[0] + l[1]*l[1] + l[2]*l[2] + 1e-8f);
  size_t base = (size_t)i * COUT + HH * CC;  // after o block (192)
  cat[base + 0 * (HH*PV) + pp] = l[0];
  cat[base + 1 * (HH*PV) + pp] = l[1];
  cat[base + 2 * (HH*PV) + pp] = l[2];
  cat[base + 3 * (HH*PV) + pp] = nrm;
}

// ---------------- host-side orchestration ----------------
extern "C" void kernel_launch(void* const* d_in, const int* in_sizes, int n_in,
                              void* d_out, int out_size, void* d_ws, size_t ws_size,
                              hipStream_t stream) {
  (void)in_sizes; (void)n_in; (void)out_size; (void)ws_size;
  const float* s     = (const float*)d_in[0];
  const float* z     = (const float*)d_in[1];
  const float* rot   = (const float*)d_in[2];
  const float* trans = (const float*)d_in[3];
  const float* mask  = (const float*)d_in[4];
  const float* Wq    = (const float*)d_in[5];
  const float* bq    = (const float*)d_in[6];
  const float* Wkv   = (const float*)d_in[7];
  const float* bkv   = (const float*)d_in[8];
  const float* Wqp   = (const float*)d_in[9];
  const float* bqp   = (const float*)d_in[10];
  const float* Wkvp  = (const float*)d_in[11];
  const float* bkvp  = (const float*)d_in[12];
  const float* Wb    = (const float*)d_in[13];
  const float* bb    = (const float*)d_in[14];
  const float* hwts  = (const float*)d_in[15];
  const float* Wout  = (const float*)d_in[16];
  const float* bout  = (const float*)d_in[17];
  float* out = (float*)d_out;

  char* ws = (char*)d_ws;
  auto alloc = [&](size_t bytes) {
    char* p = ws;
    ws += (bytes + 255) & ~(size_t)255;
    return p;
  };
  float*     q_ws     = (float*)alloc((size_t)NN * HH * CC * 4);          // [i][h*C]
  float*     kv_ws    = (float*)alloc((size_t)NN * 2 * HH * CC * 4);      // [j][k|v]
  float*     qp_proj  = (float*)alloc((size_t)NN * HH * PQ * 3 * 4);
  float*     kvp_proj = (float*)alloc((size_t)NN * HH * 12 * 3 * 4);
  float*     qpts     = (float*)alloc((size_t)NN * HH * 12 * 4);
  float*     kpts     = (float*)alloc((size_t)NN * HH * 12 * 4);
  float*     vpts     = (float*)alloc((size_t)NN * HH * 24 * 4);
  float*     qn_ws    = (float*)alloc((size_t)HH * NN * 4);
  float*     kn_ws    = (float*)alloc((size_t)HH * NN * 4);
  float*     bias_ws  = (float*)alloc((size_t)HH * NN * NN * 4);
  _Float16*  a_ws     = (_Float16*)alloc((size_t)16 * NN * NN * 2);
  float*     optraw   = (float*)alloc((size_t)NN * HH * 24 * 4);
  float*     cat      = (float*)alloc((size_t)NN * COUT * 4);

  // 1) linear projections (WMMA GEMMs, K = 384)
  hipLaunchKernelGGL(gemm_wmma, dim3(NN/16, (HH*CC)/16),     dim3(32), 0, stream,
                     s, CS, Wq,   CS, bq,   q_ws,     HH*CC,     CS);
  hipLaunchKernelGGL(gemm_wmma, dim3(NN/16, (2*HH*CC)/16),   dim3(32), 0, stream,
                     s, CS, Wkv,  CS, bkv,  kv_ws,    2*HH*CC,   CS);
  hipLaunchKernelGGL(gemm_wmma, dim3(NN/16, (HH*PQ*3)/16),   dim3(32), 0, stream,
                     s, CS, Wqp,  CS, bqp,  qp_proj,  HH*PQ*3,   CS);
  hipLaunchKernelGGL(gemm_wmma, dim3(NN/16, (HH*12*3)/16),   dim3(32), 0, stream,
                     s, CS, Wkvp, CS, bkvp, kvp_proj, HH*12*3,   CS);

  // 2) apply frames, point norms
  hipLaunchKernelGGL(frames_kernel, dim3((NN*HH + 127)/128), dim3(128), 0, stream,
                     qp_proj, kvp_proj, rot, trans, qpts, kpts, vpts, qn_ws, kn_ws);

  // 3) pair bias (first z stream, WMMA)
  hipLaunchKernelGGL(bias_kernel, dim3((NN*NN)/16), dim3(32), 0, stream,
                     z, Wb, bb, bias_ws);

  // 4) fused attention (logits chain -> softmax -> o / o_pt, store a)
  hipLaunchKernelGGL(attn_kernel, dim3(HH, NN/16), dim3(32), 0, stream,
                     q_ws, kv_ws, qpts, kpts, vpts, qn_ws, kn_ws, bias_ws,
                     mask, hwts, a_ws, cat, optraw);

  // 5) o_pair (second z stream, TDM double-buffered into LDS, WMMA)
  hipLaunchKernelGGL(opair_kernel, dim3(NN), dim3(32), 0, stream, a_ws, z, cat);

  // 6) finalize points into concat
  hipLaunchKernelGGL(finalize_pts_kernel, dim3((NN*HH*PV + 127)/128), dim3(128), 0, stream,
                     optraw, rot, trans, cat);

  // 7) output projection: out[N, CS] = cat @ Wout^T + bout  (K = 2112)
  hipLaunchKernelGGL(gemm_wmma, dim3(NN/16, CS/16), dim3(32), 0, stream,
                     cat, COUT, Wout, COUT, bout, out, CS, COUT);
}
